// TensorNetInteraction_45930380264226
// MI455X (gfx1250) — compile-verified, matrix operand order checked by reading, not run on hardware
//
#include <hip/hip_runtime.h>
#include <hip/hip_bf16.h>
#include <math.h>

typedef __bf16 bf16_t;
typedef __attribute__((ext_vector_type(16))) __bf16 v16bf;
typedef __attribute__((ext_vector_type(8)))  float  v8f;

#define FEAT 64
#define NRBF_K 32

__device__ __forceinline__ float fast_rcp(float x) { return __builtin_amdgcn_rcpf(x); }
__device__ __forceinline__ float silu_f(float x) { return x * fast_rcp(1.0f + __expf(-x)); }

__device__ __forceinline__ v8f wmma_bf16(v16bf a, v16bf b, v8f c) {
  // D = A(16x32 bf16) * B(32x16 bf16) + C(16x16 f32)
  return __builtin_amdgcn_wmma_f32_16x16x32_bf16(false, a, false, b, (short)0, c, false, false);
}

// ISA 7.12.2 K index for 16-bit A/B fragments: VGPR v<4: K=kh*8+2v, v>=4: K=16+kh*8+2(v-4)
__device__ __forceinline__ int frag_k0(int v, int kh) {
  return ((v & 4) << 2) + kh * 8 + ((v & 3) << 1);
}

// A-fragment: 16(M) x 32(K) bf16, source row-major with leading dim ldm (vectorizes to 2x ds_load_b128)
__device__ __forceinline__ v16bf load_a_frag(const bf16_t* base, int ldm, int lane) {
  int m  = lane & 15;
  int kh = (lane >> 4) & 1;
  const bf16_t* r = base + m * ldm;
  v16bf a;
#pragma unroll
  for (int v = 0; v < 8; ++v) {
    int k0 = frag_k0(v, kh);
    a[2 * v]     = r[k0];
    a[2 * v + 1] = r[k0 + 1];
  }
  return a;
}

// B-fragment from pre-swizzled fragment-major storage: lane's 16 bf16 are contiguous (32B aligned)
__device__ __forceinline__ v16bf load_b_packed(const bf16_t* tile_base, int lane) {
  return *(const v16bf*)(tile_base + lane * 16);
}

// ---------------- irreducible 3x3 <-> 10-component helpers ----------------
// comps: [dm, a01, a02, a12, s00, s11, s22, s01, s02, s12]
__device__ __forceinline__ void comps_to_full(const float* c, float* t) {
  float dm = c[0], a01 = c[1], a02 = c[2], a12 = c[3];
  float s00 = c[4], s11 = c[5], s22 = c[6], s01 = c[7], s02 = c[8], s12 = c[9];
  t[0] = dm + s00;  t[1] = a01 + s01;  t[2] = a02 + s02;
  t[3] = -a01 + s01; t[4] = dm + s11;  t[5] = a12 + s12;
  t[6] = -a02 + s02; t[7] = -a12 + s12; t[8] = dm + s22;
}

__device__ __forceinline__ void full_to_comps(const float* t, float scale, float* c) {
  float dm = (t[0] + t[4] + t[8]) * (1.0f / 3.0f);
  c[0] = dm * scale;
  c[1] = 0.5f * (t[1] - t[3]) * scale;
  c[2] = 0.5f * (t[2] - t[6]) * scale;
  c[3] = 0.5f * (t[5] - t[7]) * scale;
  c[4] = (t[0] - dm) * scale;
  c[5] = (t[4] - dm) * scale;
  c[6] = (t[8] - dm) * scale;
  c[7] = 0.5f * (t[1] + t[3]) * scale;
  c[8] = 0.5f * (t[2] + t[6]) * scale;
  c[9] = 0.5f * (t[5] + t[7]) * scale;
}

// ---------------- kernel: weight prep: fp32 -> bf16, swizzled to WMMA B-fragment layout ----------------
// dst[(kc*NT+nt)*512 + lane*16 + j] = W[ (nt*16 + (lane&15)) * Ktot + kc*32 + frag_k0(j>>1, lane>>4) + (j&1) ]
__device__ __forceinline__ void pack_one(const float* __restrict__ W, bf16_t* __restrict__ dst,
                                         int i, int NT, int Ktot) {
  int tile = i >> 9;
  int r = i & 511;
  int lane = r >> 4;
  int j = r & 15;
  int kc = tile / NT, nt = tile % NT;
  int n = lane & 15, kh = lane >> 4;
  int k = kc * 32 + frag_k0(j >> 1, kh) + (j & 1);
  dst[i] = (bf16_t)W[(nt * 16 + n) * Ktot + k];
}

__global__ void kprep(const float* __restrict__ W0, const float* __restrict__ W1,
                      const float* __restrict__ W2, const float* __restrict__ Wt,
                      bf16_t* B0, bf16_t* B1, bf16_t* B2, bf16_t* Wtb) {
  int i = blockIdx.x * 256 + threadIdx.x;
  if (i < 32 * 64)   pack_one(W0, B0, i, 4, 32);    // K=32,  N=64
  if (i < 64 * 128)  pack_one(W1, B1, i, 8, 64);    // K=64,  N=128
  if (i < 128 * 192) pack_one(W2, B2, i, 12, 128);  // K=128, N=192
  if (i < 6 * 4096) {                                // 6x (K=64, N=64), source Wt[w][g][f]
    int w = i >> 12;
    pack_one(Wt + w * 4096, Wtb + w * 4096, i & 4095, 4, 64);
  }
}

// ---------------- kernel: fused 3-layer edge MLP with bf16 WMMA ----------------
// per block: 128 edges, 8 waves; activations staged in LDS as bf16
__global__ void __launch_bounds__(256)
k1_edge_mlp(const float* __restrict__ rbf, const float* __restrict__ dij,
            const bf16_t* __restrict__ B0, const bf16_t* __restrict__ B1,
            const bf16_t* __restrict__ B2,
            const float* __restrict__ b0, const float* __restrict__ b1,
            const float* __restrict__ b2,
            bf16_t* __restrict__ rfv_out, int E) {
  __shared__ bf16_t act0[128 * 32];   //  8 KB
  __shared__ bf16_t act1[128 * 64];   // 16 KB
  __shared__ bf16_t act2[128 * 128];  // 32 KB
  __shared__ float  ldsC[128];

  int tid = threadIdx.x;
  int wave = tid >> 5, lane = tid & 31;
  long eBase = (long)blockIdx.x * 128;

  // stage input + cutoff
  for (int i = tid; i < 128 * 32; i += 256) {
    int m = i >> 5, k = i & 31;
    long e = eBase + m;
    act0[m * 32 + k] = (bf16_t)((e < E) ? rbf[e * NRBF_K + k] : 0.0f);
  }
  for (int i = tid; i < 128; i += 256) {
    long e = eBase + i;
    float d = (e < E) ? dij[e] : 2.0f;
    ldsC[i] = (d < 1.0f) ? 0.5f * (__cosf(3.14159265358979f * d) + 1.0f) : 0.0f;
  }
  __syncthreads();

  // ---- L0: [16x32] @ [32x64] ----
  {
    v16bf a = load_a_frag(&act0[wave * 16 * 32], 32, lane);
    int n0 = lane & 15, mh = (lane >> 4) & 1;
#pragma unroll
    for (int nt = 0; nt < 4; ++nt) {
      v16bf b = load_b_packed(B0 + nt * 512, lane);
      v8f c = {};
      c = wmma_bf16(a, b, c);
      int n = nt * 16 + n0;
      float bias = b0[n];
#pragma unroll
      for (int v = 0; v < 8; ++v) {
        int m = wave * 16 + v + mh * 8;
        act1[m * 64 + n] = (bf16_t)silu_f(c[v] + bias);
      }
    }
  }
  __syncthreads();

  // ---- L1: [16x64] @ [64x128] ----
  {
    v16bf a0 = load_a_frag(&act1[wave * 16 * 64], 64, lane);
    v16bf a1 = load_a_frag(&act1[wave * 16 * 64 + 32], 64, lane);
    int n0 = lane & 15, mh = (lane >> 4) & 1;
#pragma unroll
    for (int nt = 0; nt < 8; ++nt) {
      v16bf bb0 = load_b_packed(B1 + (0 * 8 + nt) * 512, lane);
      v16bf bb1 = load_b_packed(B1 + (1 * 8 + nt) * 512, lane);
      v8f c = {};
      c = wmma_bf16(a0, bb0, c);
      c = wmma_bf16(a1, bb1, c);
      int n = nt * 16 + n0;
      float bias = b1[n];
#pragma unroll
      for (int v = 0; v < 8; ++v) {
        int m = wave * 16 + v + mh * 8;
        act2[m * 128 + n] = (bf16_t)silu_f(c[v] + bias);
      }
    }
  }
  __syncthreads();

  // ---- L2: [16x128] @ [128x192], apply cutoff, store ----
  {
    v16bf a[4];
#pragma unroll
    for (int kc = 0; kc < 4; ++kc)
      a[kc] = load_a_frag(&act2[wave * 16 * 128 + kc * 32], 128, lane);
    int n0 = lane & 15, mh = (lane >> 4) & 1;
#pragma unroll
    for (int nt = 0; nt < 12; ++nt) {
      v8f c = {};
#pragma unroll
      for (int kc = 0; kc < 4; ++kc) {
        v16bf b = load_b_packed(B2 + (kc * 12 + nt) * 512, lane);
        c = wmma_bf16(a[kc], b, c);
      }
      int n = nt * 16 + n0;
      float bias = b2[n];
#pragma unroll
      for (int v = 0; v < 8; ++v) {
        int m = wave * 16 + v + mh * 8;
        long e = eBase + m;
        if (e < E) {
          float x = silu_f(c[v] + bias) * ldsC[m];
          rfv_out[e * 192 + n] = (bf16_t)x;
        }
      }
    }
  }
}

// ---------------- kernel: normalize X, write irreducible comps ----------------
__global__ void k2a(const float* __restrict__ X, float* __restrict__ cmpX, long NF) {
  long i = (long)blockIdx.x * 256 + threadIdx.x;
  if (i >= NF) return;
  const float* x = X + i * 9;
  float t[9], s2 = 0.0f;
#pragma unroll
  for (int j = 0; j < 9; ++j) { t[j] = x[j]; s2 += t[j] * t[j]; }
  float inv = fast_rcp(s2 + 1.0f);
  float c[10];
  full_to_comps(t, inv, c);
  float* o = cmpX + i * 10;
#pragma unroll
  for (int j = 0; j < 10; ++j) o[j] = c[j];
}

// ---------------- kernel: channel-mixing linears via WMMA (comp-wise GEMM) ----------------
// out[n,g,k] = sum_f Wsel(k)[g,f] * in[n,f,k]; Wtb holds 3 mats pre-swizzled, 4096 elems each.
__global__ void __launch_bounds__(256)
kmix(const float* __restrict__ cin, const bf16_t* __restrict__ Wtb,
     float* __restrict__ cout, int N) {
  __shared__ bf16_t ldsA[10 * 16 * 64];  // 20 KB: per comp k, [16 atoms][64 f]
  int tid = threadIdx.x;
  int wave = tid >> 5, lane = tid & 31;
  long nBase = (long)blockIdx.x * 16;

  for (int i = tid; i < 10 * 16 * 64; i += 256) {
    int m = i / 640; int r = i % 640; int f = r / 10; int k = r % 10;
    long n = nBase + m;
    ldsA[k * 1024 + m * 64 + f] = (bf16_t)((n < N) ? cin[n * 640 + f * 10 + k] : 0.0f);
  }
  __syncthreads();

  int n0 = lane & 15, mh = (lane >> 4) & 1;
#pragma unroll
  for (int i = 0; i < 5; ++i) {
    int tile = wave * 5 + i;            // 0..39 = 10 comps x 4 g-tiles
    int k = tile >> 2, gt = tile & 3;
    int wsel = (k == 0) ? 0 : ((k < 4) ? 1 : 2);
    const bf16_t* W = Wtb + wsel * 4096;
    const bf16_t* A = &ldsA[k * 1024];
    v16bf a0 = load_a_frag(A, 64, lane);
    v16bf a1 = load_a_frag(A + 32, 64, lane);
    v16bf bb0 = load_b_packed(W + (0 * 4 + gt) * 512, lane);
    v16bf bb1 = load_b_packed(W + (1 * 4 + gt) * 512, lane);
    v8f c = {};
    c = wmma_bf16(a0, bb0, c);
    c = wmma_bf16(a1, bb1, c);
    int g = gt * 16 + n0;
#pragma unroll
    for (int v = 0; v < 8; ++v) {
      long n = nBase + v + mh * 8;
      if (n < N) cout[n * 640 + g * 10 + k] = c[v];
    }
  }
}

// ---------------- kernel: zero scratch ----------------
__global__ void kzero(float* p, long count) {
  long i = (long)blockIdx.x * 256 + threadIdx.x;
  long stride = (long)gridDim.x * 256;
  for (; i < count; i += stride) p[i] = 0.0f;
}

// ---------------- kernel: message passing (gather comps at src, atomic scatter to dst) ----------------
__global__ void k3_message(const int* __restrict__ pair, const bf16_t* __restrict__ rfv,
                           const float* __restrict__ cm, float* __restrict__ msg, int E) {
  long t = (long)blockIdx.x * 256 + threadIdx.x;
  int f = (int)(t & 63);
  long e = t >> 6;
  if (e >= E) return;
  int dst = pair[e];
  int src = pair[(long)E + e];
  float r0 = (float)rfv[e * 192 + f * 3 + 0];
  float r1 = (float)rfv[e * 192 + f * 3 + 1];
  float r2 = (float)rfv[e * 192 + f * 3 + 2];
  const float* c = cm + ((long)src * FEAT + f) * 10;
  float dm = c[0], a01 = c[1], a02 = c[2], a12 = c[3];
  float s00 = c[4], s11 = c[5], s22 = c[6], s01 = c[7], s02 = c[8], s12 = c[9];
  float* o = msg + ((long)dst * FEAT + f) * 9;
  atomicAdd(&o[0], r0 * dm + r2 * s00);
  atomicAdd(&o[1],  r1 * a01 + r2 * s01);
  atomicAdd(&o[2],  r1 * a02 + r2 * s02);
  atomicAdd(&o[3], -r1 * a01 + r2 * s01);
  atomicAdd(&o[4], r0 * dm + r2 * s11);
  atomicAdd(&o[5],  r1 * a12 + r2 * s12);
  atomicAdd(&o[6], -r1 * a02 + r2 * s02);
  atomicAdd(&o[7], -r1 * a12 + r2 * s12);
  atomicAdd(&o[8], r0 * dm + r2 * s22);
}

// ---------------- kernel: O(3) stage: q*(msg@Y + Y@msg), normalize, comps ----------------
__global__ void k4a(const float* __restrict__ msg, const float* __restrict__ cm,
                    const float* __restrict__ charges, float* __restrict__ cmp2, long NF) {
  long i = (long)blockIdx.x * 256 + threadIdx.x;
  if (i >= NF) return;
  long n = i >> 6;
  float Y[9];
  comps_to_full(cm + i * 10, Y);
  const float* Mp = msg + i * 9;
  float M[9];
#pragma unroll
  for (int j = 0; j < 9; ++j) M[j] = Mp[j];
  float q = 1.0f + 0.1f * charges[n];
  float B[9];
#pragma unroll
  for (int r = 0; r < 3; ++r)
#pragma unroll
    for (int cc = 0; cc < 3; ++cc) {
      float acc = 0.0f;
#pragma unroll
      for (int k = 0; k < 3; ++k)
        acc += M[r * 3 + k] * Y[k * 3 + cc] + Y[r * 3 + k] * M[k * 3 + cc];
      B[r * 3 + cc] = q * acc;
    }
  float s2 = 0.0f;
#pragma unroll
  for (int j = 0; j < 9; ++j) s2 += B[j] * B[j];
  float inv = fast_rcp(s2 + 1.0f);
  float c[10];
  full_to_comps(B, inv, c);
  float* o = cmp2 + i * 10;
#pragma unroll
  for (int j = 0; j < 10; ++j) o[j] = c[j];
}

// ---------------- kernel: final update X + dX + q*dX@dX ----------------
__global__ void k4c(const float* __restrict__ cmpX, const float* __restrict__ cmpD,
                    const float* __restrict__ charges, float* __restrict__ out, long NF) {
  long i = (long)blockIdx.x * 256 + threadIdx.x;
  if (i >= NF) return;
  long n = i >> 6;
  float Xn[9], dX[9];
  comps_to_full(cmpX + i * 10, Xn);
  comps_to_full(cmpD + i * 10, dX);
  float q = 1.0f + 0.1f * charges[n];
  float* o = out + i * 9;
#pragma unroll
  for (int r = 0; r < 3; ++r)
#pragma unroll
    for (int cc = 0; cc < 3; ++cc) {
      float acc = 0.0f;
#pragma unroll
      for (int k = 0; k < 3; ++k) acc += dX[r * 3 + k] * dX[k * 3 + cc];
      o[r * 3 + cc] = Xn[r * 3 + cc] + dX[r * 3 + cc] + q * acc;
    }
}

extern "C" void kernel_launch(void* const* d_in, const int* in_sizes, int n_in,
                              void* d_out, int out_size, void* d_ws, size_t ws_size,
                              hipStream_t stream) {
  const float* X       = (const float*)d_in[0];
  const int*   pair    = (const int*)d_in[1];
  const float* dij     = (const float*)d_in[2];
  const float* rbf     = (const float*)d_in[3];
  const float* charges = (const float*)d_in[4];
  const float* W0 = (const float*)d_in[5];
  const float* b0 = (const float*)d_in[6];
  const float* W1 = (const float*)d_in[7];
  const float* b1 = (const float*)d_in[8];
  const float* W2 = (const float*)d_in[9];
  const float* b2 = (const float*)d_in[10];
  const float* Wt = (const float*)d_in[11];
  float* out = (float*)d_out;

  int E = in_sizes[1] / 2;
  int N = in_sizes[4];
  long NF = (long)N * FEAT;

  char* ws = (char*)d_ws;
  size_t off = 0;
  auto alloc = [&](size_t bytes) -> void* {
    void* p = ws + off;
    off = (off + bytes + 255) & ~(size_t)255;
    return p;
  };
  bf16_t* B0b  = (bf16_t*)alloc(32 * 64 * sizeof(bf16_t));
  bf16_t* B1b  = (bf16_t*)alloc(64 * 128 * sizeof(bf16_t));
  bf16_t* B2b  = (bf16_t*)alloc(128 * 192 * sizeof(bf16_t));
  bf16_t* Wtb  = (bf16_t*)alloc(6 * 4096 * sizeof(bf16_t));
  bf16_t* rfv  = (bf16_t*)alloc((size_t)E * 192 * sizeof(bf16_t));
  float*  cmpX = (float*)alloc((size_t)NF * 10 * sizeof(float));
  float*  cmpM = (float*)alloc((size_t)NF * 10 * sizeof(float));  // reused as dX comps
  float*  cmp2 = (float*)alloc((size_t)NF * 10 * sizeof(float));
  float*  msg  = (float*)alloc((size_t)NF * 9 * sizeof(float));

  kprep<<<96, 256, 0, stream>>>(W0, W1, W2, Wt, B0b, B1b, B2b, Wtb);
  k1_edge_mlp<<<(E + 127) / 128, 256, 0, stream>>>(rbf, dij, B0b, B1b, B2b, b0, b1, b2, rfv, E);
  k2a<<<(unsigned)((NF + 255) / 256), 256, 0, stream>>>(X, cmpX, NF);
  kmix<<<(N + 15) / 16, 256, 0, stream>>>(cmpX, Wtb, cmpM, N);
  kzero<<<2048, 256, 0, stream>>>(msg, NF * 9);
  k3_message<<<(unsigned)(((long)E * FEAT + 255) / 256), 256, 0, stream>>>(pair, rfv, cmpM, msg, E);
  k4a<<<(unsigned)((NF + 255) / 256), 256, 0, stream>>>(msg, cmpM, charges, cmp2, NF);
  kmix<<<(N + 15) / 16, 256, 0, stream>>>(cmp2, Wtb + 3 * 4096, cmpM, N);  // cmpM now holds dX comps
  k4c<<<(unsigned)((NF + 255) / 256), 256, 0, stream>>>(cmpX, cmpM, charges, out, NF);
}
